// APLHelixNetwork_15925738734068
// MI455X (gfx1250) — matrix operand order repaired and checked by reading, not compiled, not run on hardware
//
#include <hip/hip_runtime.h>
#include <hip/hip_bf16.h>
#include <math.h>
#include <stdint.h>

#define NOSC 60
#define NPAD 64
#define BATCH 8192
#define DIN 512
#define DOUT 256
#define NLAYER 4
#define NSTEP 10

typedef _Float16 h16;
typedef __attribute__((ext_vector_type(16))) _Float16 v16h;
typedef __attribute__((ext_vector_type(8)))  _Float16 v8h;
typedef __attribute__((ext_vector_type(8)))  float    v8f;

// ---------------------------------------------------------------------------
// WMMA fragment helpers (wave32; A/B use contiguous k order, C/D per CDNA5 ISA 7.12.2)
// A (16x32 f16): lanes 0-15 -> M=lane, K chunks {kb, kb+8} with kb = kc*32 + 0
//                lanes 16-31 -> M=lane-16, kb = kc*32 + 16
// B (32x16 f16): lane n holds column n: K = kc*32 + [0..15] (lanes 0-15)
//                                       K = kc*32 + [16..31] (lanes 16-31)
// C/D (16x16 f32): vgpr v, lane l -> row = v + 8*(l>=16), col = l & 15
// ---------------------------------------------------------------------------
__device__ __forceinline__ v16h load_a_frag(const h16* base, int stride, int kc) {
    const int lane = threadIdx.x & 31;
    const int hf = lane >> 4;
    const int r = lane & 15;
    const int kb0 = kc * 32 + hf * 16;
    const int kb1 = kb0 + 8;
    v8h lo = *(const v8h*)(base + r * stride + kb0);
    v8h hi = *(const v8h*)(base + r * stride + kb1);
    v16h f;
#pragma unroll
    for (int i = 0; i < 8; ++i) { f[i] = lo[i]; f[i + 8] = hi[i]; }
    return f;
}

__device__ __forceinline__ v16h load_b_frag(const h16* bt, int stride, int nt, int kc) {
    const int lane = threadIdx.x & 31;
    const int hf = lane >> 4;
    const int r = lane & 15;
    const int kb = kc * 32 + hf * 16;
    return *(const v16h*)(bt + (nt * 16 + r) * stride + kb);
}

__device__ __forceinline__ v8f wmma_f16(v16h a, v16h b, v8f c) {
    return __builtin_amdgcn_wmma_f32_16x16x32_f16(false, a, false, b,
                                                  (short)0, c, false, false);
}

__device__ __forceinline__ float gelu_exact(float x) {
    return 0.5f * x * (1.f + erff(x * 0.7071067811865475f));
}

// wrap to principal value (-pi, pi]; equivalent to atan2(sin x, cos x)
__device__ __forceinline__ float wrap_pi(float x) {
    return x - 6.28318530717958647693f * rintf(x * 0.15915494309189533577f);
}

// ---------------------------------------------------------------------------
// Prep: transpose + f32->f16 convert weight copies, zero z accumulators
// ---------------------------------------------------------------------------
__global__ void prep_kernel(const float* __restrict__ W_enc,
                            const float* __restrict__ W_out,
                            const float* __restrict__ K,
                            h16* __restrict__ wenc_t,   // [64][512]
                            h16* __restrict__ wout_t,   // [256][128]
                            h16* __restrict__ Kt,       // [4][64][64]
                            float* __restrict__ zsum) { // [4]
    const int i = blockIdx.x * blockDim.x + threadIdx.x;
    if (i < 64 * DIN) {
        const int n = i >> 9, k = i & (DIN - 1);
        wenc_t[i] = (h16)((n < NOSC) ? W_enc[k * NOSC + n] : 0.f);
    }
    if (i < DOUT * 128) {
        const int n = i >> 7, k = i & 127;
        wout_t[i] = (h16)((k < 2 * NOSC) ? W_out[k * DOUT + n] : 0.f);
    }
    if (i < NLAYER * NPAD * NPAD) {
        const int l = i >> 12, rem = i & 4095, r = rem >> 6, c = rem & 63;
        Kt[i] = (h16)((r < NOSC && c < NOSC) ? K[(l * NOSC + r) * NOSC + c] : 0.f);
    }
    if (i < NLAYER) zsum[i] = 0.f;
}

// ---------------------------------------------------------------------------
// Encoder: theta = pi * tanh(x @ W_enc + b_enc).  [8192,512]x[512,60] WMMA.
// Block = 128 thr (4 waves), 16 batch rows; wave w computes n-tile w.
// ---------------------------------------------------------------------------
__global__ void encode_kernel(const float* __restrict__ x,
                              const h16* __restrict__ wenc_t,
                              const float* __restrict__ b_enc,
                              float* __restrict__ theta) {
    __shared__ __align__(16) h16 a_s[16][DIN];
    const int t = threadIdx.x;
    const int mbase = blockIdx.x * 16;
    for (int i = t; i < 16 * DIN; i += 128) {
        const int r = i >> 9, c = i & (DIN - 1);
        a_s[r][c] = (h16)x[(size_t)(mbase + r) * DIN + c];
    }
    __syncthreads();
    const int wave = t >> 5, lane = t & 31, hf = lane >> 4, r = lane & 15;
    const int nt = wave;
    v8f acc = {};
#pragma unroll 4
    for (int kc = 0; kc < DIN / 32; ++kc) {
        v16h a = load_a_frag(&a_s[0][0], DIN, kc);
        v16h b = load_b_frag(wenc_t, DIN, nt, kc);
        acc = wmma_f16(a, b, acc);
    }
    const int col = nt * 16 + r;
    if (col < NOSC) {
        const float bb = b_enc[col];
#pragma unroll
        for (int v = 0; v < 8; ++v) {
            const int m = v + hf * 8;
            theta[(size_t)(mbase + m) * NOSC + col] =
                3.14159265358979323846f * tanhf(acc[v] + bb);
        }
    }
}

// ---------------------------------------------------------------------------
// Coherence: coh[b] = |<e^{i theta}>| per row; zsum += coh (for z = mean coh).
// One wave per row (wave32 -> 2 elements/lane), 4 rows per block.
// ---------------------------------------------------------------------------
__global__ void coh_kernel(const float* __restrict__ theta,
                           float* __restrict__ coh,
                           float* __restrict__ zsum) {
    __shared__ float part[4];
    const int t = threadIdx.x;
    const int wave = t >> 5, lane = t & 31;
    const int row = blockIdx.x * 4 + wave;
    float th0 = theta[(size_t)row * NOSC + lane];
    float cs = cosf(th0), sn = sinf(th0);
    if (lane < NOSC - 32) {
        float th1 = theta[(size_t)row * NOSC + lane + 32];
        cs += cosf(th1); sn += sinf(th1);
    }
#pragma unroll
    for (int off = 16; off >= 1; off >>= 1) {
        cs += __shfl_xor(cs, off, 32);
        sn += __shfl_xor(sn, off, 32);
    }
    const float c = sqrtf((cs / NOSC) * (cs / NOSC) + (sn / NOSC) * (sn / NOSC));
    if (lane == 0) { coh[row] = c; part[wave] = c; }
    __syncthreads();
    if (t == 0) atomicAdd(zsum, part[0] + part[1] + part[2] + part[3]);
}

// ---------------------------------------------------------------------------
// Controller: only logits[0] feeds argmax -> evaluate MLP for row 0 only.
// Single block, 64 threads. Emits per-layer scalars: fac, coef, dmask, omdt[].
// ---------------------------------------------------------------------------
__global__ void controller_kernel(int l,
                                  const float* __restrict__ theta,  // row 0
                                  const float* __restrict__ coh,
                                  const float* __restrict__ zsum_l,
                                  const float* __restrict__ omega,
                                  const float* __restrict__ K_global,
                                  const float* __restrict__ apl_K_mod,
                                  const float* __restrict__ apl_omega_mod,
                                  const float* __restrict__ W1,
                                  const float* __restrict__ b1,
                                  const float* __restrict__ W2,
                                  const float* __restrict__ b2,
                                  const float* __restrict__ W3,
                                  const float* __restrict__ b3,
                                  const float* __restrict__ prior,
                                  float* __restrict__ omdt,   // [64]
                                  float* __restrict__ scal) { // fac, coef, dmask
    __shared__ float feats[64], hh1[64], hh2[64];
    __shared__ int s_idx;
    const int t = threadIdx.x;
    const float ZCv = 0.8660254037844386468f;
    const float z = zsum_l[0] * (1.f / (float)BATCH);
    const float bounds[8] = {0.1f, 0.2f, 0.4f, 0.6f, 0.75f, ZCv, 0.92f, 0.97f};
    int tier = 0;
#pragma unroll
    for (int k = 0; k < 8; ++k) tier += (z >= bounds[k]) ? 1 : 0;
    const float dsneg = expf(-36.f * (z - ZCv) * (z - ZCv));

    if (t < NOSC)           feats[t] = cosf(theta[t]);
    else if (t == NOSC)     feats[t] = z;
    else if (t == NOSC + 1) feats[t] = coh[0];
    else if (t == NOSC + 2) feats[t] = dsneg;
    else                    feats[t] = ((float)tier + 1.f) / 9.f;
    __syncthreads();

    const float* W1l = W1 + l * 64 * 64;
    const float* W2l = W2 + l * 64 * 64;
    const float* W3l = W3 + l * 64 * 6;
    float a = b1[l * 64 + t];
    for (int i = 0; i < 64; ++i) a += feats[i] * W1l[i * 64 + t];
    hh1[t] = gelu_exact(a);
    __syncthreads();
    a = b2[l * 64 + t];
    for (int i = 0; i < 64; ++i) a += hh1[i] * W2l[i * 64 + t];
    hh2[t] = gelu_exact(a);
    __syncthreads();

    if (t == 0) {
        const bool legal[9][6] = {
            {0,0,0,1,1,1},{1,0,1,0,1,1},{1,1,1,0,1,1},{0,1,0,1,1,1},
            {1,1,1,1,1,1},{0,1,0,1,1,1},{0,1,0,1,0,0},{0,1,1,1,0,0},{0,1,1,1,0,0}};
        float best = -1e30f; int bi = 0;
        for (int j = 0; j < 6; ++j) {
            if (!legal[tier][j]) continue;
            float lg = b3[l * 6 + j] + prior[l * 6 + j];
            for (int i = 0; i < 64; ++i) lg += hh2[i] * W3l[i * 6 + j];
            if (lg > best) { best = lg; bi = j; }
        }
        s_idx = bi;
    }
    __syncthreads();
    const int idx = s_idx;
    const float parity[6] = {1.f, -1.f, 1.f, 1.f, -1.f, -1.f};
    const float par = parity[idx];
    if (t == 0) {
        const float kscale = 1.f + par * (apl_K_mod[l * 6 + idx] * z) * 0.5f;
        scal[0] = 0.1f * (K_global[l] / (float)NOSC) * kscale;        // DT*kg/N*kscale
        scal[1] = (idx == 1) ? 0.1f : ((idx == 5) ? -0.1f : 0.f);     // coef
        scal[2] = (idx == 4) ? 1.f : 0.f;                             // dmask
    }
    if (t < NOSC)
        omdt[t] = 0.1f * (omega[l * NOSC + t] + apl_omega_mod[l * 6 + idx] * par);
}

// ---------------------------------------------------------------------------
// Fused Kuramoto layer: pre-step + 10 steps entirely in LDS.
// sum_j K[i,j] sin(th_j - th_i) = cos(th_i)*(S@K)_i - sin(th_i)*(C@K)_i
// -> two [16,64]x[64,64] f16 WMMA GEMMs per step per wave.
// Single-wave workgroup (barrier -> S_NOP) owning 16 batch rows;
// sin/cos double-buffered so WMMA A-fragments never race their updates.
// K tile staged via CDNA5 async global->LDS DMA (ASYNCcnt).
// ---------------------------------------------------------------------------
__global__ void kuramoto_kernel(float* __restrict__ theta,
                                const h16* __restrict__ Kt,     // [64][64] layer slice
                                const float* __restrict__ omdt,
                                const float* __restrict__ scal,
                                const float* __restrict__ coh,
                                const float* __restrict__ noise) { // layer slice
    __shared__ __align__(16) float th_s[16][NPAD];
    __shared__ __align__(16) h16 sn_s[2][16][NPAD];
    __shared__ __align__(16) h16 cs_s[2][16][NPAD];
    __shared__ __align__(16) h16 K_s[NPAD][NPAD];
    __shared__ float rowmean[16];

    const int t = threadIdx.x;          // 0..31, one wave32
    const int rbase = blockIdx.x * 16;
    const float fac = scal[0], coef = scal[1], dmask = scal[2];

    // Async DMA of the K tile (8 KB, L2-resident) into LDS; overlaps with the
    // theta load + pre-step math below. Drained with s_wait_asynccnt.
    for (int c = t; c < (NPAD * NPAD) / 8; c += 32) {
        unsigned lds_addr = (unsigned)(uintptr_t)(((h16*)K_s) + c * 8);
        const h16* gp = Kt + c * 8;
        asm volatile("global_load_async_to_lds_b128 %0, %1, off"
                     :: "v"(lds_addr), "v"(gp) : "memory");
    }

    for (int i = t; i < 16 * NPAD; i += 32) {
        const int r = i >> 6, c = i & 63;
        th_s[r][c] = (c < NOSC) ? theta[(size_t)(rbase + r) * NOSC + c] : 0.f;
    }
    __syncthreads();
    if (t < 16) {
        float s = 0.f;
        for (int c = 0; c < NOSC; ++c) s += th_s[t][c];
        rowmean[t] = s * (1.f / (float)NOSC);
    }
    __syncthreads();
    // pre-step ('+'/'-' shear + 'div' noise) and initial sin/cos
    for (int i = t; i < 16 * NPAD; i += 32) {
        const int r = i >> 6, c = i & 63;
        if (c < NOSC) {
            float th = th_s[r][c];
            th += coef * __sinf(rowmean[r] - th)
                + dmask * 0.05f * noise[(size_t)(rbase + r) * NOSC + c]
                        * (1.f - coh[rbase + r]);
            th_s[r][c] = th;
            sn_s[0][r][c] = (h16)__sinf(th);
            cs_s[0][r][c] = (h16)__cosf(th);
        } else {
            sn_s[0][r][c] = (h16)0.f; cs_s[0][r][c] = (h16)0.f;
            sn_s[1][r][c] = (h16)0.f; cs_s[1][r][c] = (h16)0.f;
        }
    }
    asm volatile("s_wait_asynccnt 0x0" ::: "memory");
    __syncthreads();

    const int hf = t >> 4, r = t & 15;
    int cur = 0;
    for (int step = 0; step < NSTEP; ++step) {
        const int nxt = cur ^ 1;
#pragma unroll
        for (int nt = 0; nt < 4; ++nt) {
            v8f accS = {};
            v8f accC = {};
#pragma unroll
            for (int kc = 0; kc < 2; ++kc) {
                v16h aS = load_a_frag(&sn_s[cur][0][0], NPAD, kc);
                v16h aC = load_a_frag(&cs_s[cur][0][0], NPAD, kc);
                v16h bK = load_b_frag(&K_s[0][0], NPAD, nt, kc);
                accS = wmma_f16(aS, bK, accS);
                accC = wmma_f16(aC, bK, accC);
            }
            const int i = nt * 16 + r;
#pragma unroll
            for (int v = 0; v < 8; ++v) {
                const int m = v + hf * 8;
                if (i < NOSC) {
                    const float smi = (float)sn_s[cur][m][i];
                    const float cmi = (float)cs_s[cur][m][i];
                    const float csum = cmi * accS[v] - smi * accC[v];
                    float thn = th_s[m][i] + omdt[i] + fac * csum;
                    const float ns = __sinf(thn), nc = __cosf(thn);
                    th_s[m][i] = wrap_pi(thn);   // == atan2(ns, nc)
                    sn_s[nxt][m][i] = (h16)ns;
                    cs_s[nxt][m][i] = (h16)nc;
                }
            }
        }
        cur = nxt;
        __syncthreads();
    }
    for (int i = t; i < 16 * NOSC; i += 32) {
        const int r2 = i / NOSC, c2 = i % NOSC;
        theta[(size_t)(rbase + r2) * NOSC + c2] = th_s[r2][c2];
    }
}

// ---------------------------------------------------------------------------
// Output: out = ([cos th | sin th] @ W_out + b_out) * coh_final.
// [8192,120]x[120,256] WMMA (K padded to 128); coh fused per 16-row tile.
// ---------------------------------------------------------------------------
__global__ void final_kernel(const float* __restrict__ theta,
                             const h16* __restrict__ wout_t,
                             const float* __restrict__ b_out,
                             float* __restrict__ out) {
    __shared__ __align__(16) h16 a_s[16][128];
    __shared__ float coh_s[16];
    const int t = threadIdx.x;
    const int mbase = blockIdx.x * 16;
    for (int i = t; i < 16 * 128; i += 128) {
        const int r = i >> 7, c = i & 127;
        float v = 0.f;
        if (c < NOSC)            v = cosf(theta[(size_t)(mbase + r) * NOSC + c]);
        else if (c < 2 * NOSC)   v = sinf(theta[(size_t)(mbase + r) * NOSC + (c - NOSC)]);
        a_s[r][c] = (h16)v;
    }
    __syncthreads();
    if (t < 16) {
        float mc = 0.f, ms = 0.f;
        for (int c = 0; c < NOSC; ++c) {
            const float th = theta[(size_t)(mbase + t) * NOSC + c];
            mc += cosf(th); ms += sinf(th);
        }
        mc *= (1.f / NOSC); ms *= (1.f / NOSC);
        coh_s[t] = sqrtf(mc * mc + ms * ms);
    }
    __syncthreads();
    const int wave = t >> 5, lane = t & 31, hf = lane >> 4, r = lane & 15;
#pragma unroll
    for (int q = 0; q < 4; ++q) {
        const int nt = wave + q * 4;   // 16 n-tiles over 4 waves
        v8f acc = {};
#pragma unroll
        for (int kc = 0; kc < 4; ++kc) {
            v16h a = load_a_frag(&a_s[0][0], 128, kc);
            v16h b = load_b_frag(wout_t, 128, nt, kc);
            acc = wmma_f16(a, b, acc);
        }
        const int col = nt * 16 + r;
        const float bb = b_out[col];
#pragma unroll
        for (int v = 0; v < 8; ++v) {
            const int m = v + hf * 8;
            out[(size_t)(mbase + m) * DOUT + col] = (acc[v] + bb) * coh_s[m];
        }
    }
}

// ---------------------------------------------------------------------------
extern "C" void kernel_launch(void* const* d_in, const int* in_sizes, int n_in,
                              void* d_out, int out_size, void* d_ws, size_t ws_size,
                              hipStream_t stream) {
    const float* x        = (const float*)d_in[0];
    const float* W_enc    = (const float*)d_in[1];
    const float* b_enc    = (const float*)d_in[2];
    const float* K        = (const float*)d_in[3];
    const float* omega    = (const float*)d_in[4];
    const float* K_global = (const float*)d_in[5];
    const float* apl_K    = (const float*)d_in[6];
    const float* apl_om   = (const float*)d_in[7];
    const float* W1 = (const float*)d_in[8];
    const float* b1 = (const float*)d_in[9];
    const float* W2 = (const float*)d_in[10];
    const float* b2 = (const float*)d_in[11];
    const float* W3 = (const float*)d_in[12];
    const float* b3 = (const float*)d_in[13];
    const float* prior = (const float*)d_in[14];
    const float* noise = (const float*)d_in[15];
    const float* W_out = (const float*)d_in[16];
    const float* b_out = (const float*)d_in[17];
    float* out = (float*)d_out;

    char* ws = (char*)d_ws;
    size_t off = 0;
    auto alloc = [&](size_t bytes) -> char* {
        char* p = ws + off;
        off += (bytes + 255) & ~(size_t)255;
        return p;
    };
    float* zsum    = (float*)alloc(NLAYER * sizeof(float));
    float* coh     = (float*)alloc(BATCH * sizeof(float));
    float* omdt    = (float*)alloc(NPAD * sizeof(float));
    float* scal    = (float*)alloc(4 * sizeof(float));
    float* theta   = (float*)alloc((size_t)BATCH * NOSC * sizeof(float));
    h16*   wenc_t  = (h16*)alloc((size_t)NPAD * DIN * sizeof(h16));
    h16*   wout_t  = (h16*)alloc((size_t)DOUT * 128 * sizeof(h16));
    h16*   Kt      = (h16*)alloc((size_t)NLAYER * NPAD * NPAD * sizeof(h16));
    (void)ws_size; (void)in_sizes; (void)n_in; (void)out_size;

    prep_kernel<<<128, 256, 0, stream>>>(W_enc, W_out, K, wenc_t, wout_t, Kt, zsum);
    encode_kernel<<<BATCH / 16, 128, 0, stream>>>(x, wenc_t, b_enc, theta);

    for (int l = 0; l < NLAYER; ++l) {
        coh_kernel<<<BATCH / 4, 128, 0, stream>>>(theta, coh, zsum + l);
        controller_kernel<<<1, 64, 0, stream>>>(l, theta, coh, zsum + l,
                                                omega, K_global, apl_K, apl_om,
                                                W1, b1, W2, b2, W3, b3, prior,
                                                omdt, scal);
        kuramoto_kernel<<<BATCH / 16, 32, 0, stream>>>(
            theta, Kt + (size_t)l * NPAD * NPAD, omdt, scal, coh,
            noise + (size_t)l * BATCH * NOSC);
    }
    final_kernel<<<BATCH / 16, 128, 0, stream>>>(theta, wout_t, b_out, out);
}
